// ZeroShotTextGNN_61787399520427
// MI455X (gfx1250) — compile-verified, matrix operand order checked by reading, not run on hardware
//
#include <hip/hip_runtime.h>
#include <hip/hip_bf16.h>
#include <math.h>

typedef __attribute__((ext_vector_type(16))) _Float16 v16h;
typedef __attribute__((ext_vector_type(8)))  _Float16 v8h;
typedef __attribute__((ext_vector_type(8)))  float    v8f;

#define Q_N   4096
#define M_N   65536
#define D_N   384
#define C_N   128
#define KT    12              // D / 32 k-chunks
#define SPLIT 64              // M splits for the fused GEMM/argmax
#define CHUNK (M_N / SPLIT)   // 1024 memory rows per chunk
#define TILES (CHUNK / 16)    // 64 16-wide tiles per chunk
#define WPB   8               // waves per block in the GEMM kernel
#define QPB   (16 * WPB)      // 128 query rows per block
#define PITCH 392             // LDS row pitch in halfs (784 B: 768 + 16 pad)
#define EPS   1e-8f
#define THRESH 0.7f

// ---------------------------------------------------------------------------
// Kernel 1: L2-normalize rows of a [rows, 384] f32 matrix into f16.
// One wave (32 lanes) per row; 4 rows per 128-thread block.
// ---------------------------------------------------------------------------
__global__ __launch_bounds__(128) void normalize_rows_f16(
    const float* __restrict__ in, _Float16* __restrict__ out, int rows) {
  const int wave = threadIdx.x >> 5;
  const int lane = threadIdx.x & 31;
  const int row  = blockIdx.x * 4 + wave;
  if (row >= rows) return;

  const float* src = in + (size_t)row * D_N;
  _Float16*    dst = out + (size_t)row * D_N;

  float xv[KT];
  float ss = 0.0f;
#pragma unroll
  for (int i = 0; i < KT; ++i) {
    float x = src[lane + i * 32];
    xv[i] = x;
    ss += x * x;
  }
#pragma unroll
  for (int m = 16; m >= 1; m >>= 1)
    ss += __shfl_xor(ss, m, 32);

  float inv = 1.0f / fmaxf(sqrtf(ss), EPS);
#pragma unroll
  for (int i = 0; i < KT; ++i)
    dst[lane + i * 32] = (_Float16)(xv[i] * inv);
}

// ---------------------------------------------------------------------------
// Kernel 2: fused cosine-sim GEMM (WMMA f16 -> f32) + running max/argmax.
// Grid: (Q/128, SPLIT), block = 256 threads = 8 waves.
// Each wave owns one 16-query block; its A operand (16 x 384) lives in 96
// VGPRs. B tiles (16 m-rows x 384) are staged through double-buffered LDS,
// loaded from global once per workgroup and consumed by all 8 waves:
// 8x reduction in global/L2 B traffic vs. per-wave streaming.
// ---------------------------------------------------------------------------
__global__ __launch_bounds__(256) void sims_argmax_wmma(
    const _Float16* __restrict__ qn, const _Float16* __restrict__ mn,
    float* __restrict__ pMax, int* __restrict__ pIdx) {
  __shared__ _Float16 Bsh[2][16 * PITCH];   // 2 x 12.25 KB

  const int tid     = threadIdx.x;
  const int wave    = tid >> 5;
  const int lane    = tid & 31;
  const int col     = lane & 15;        // operand row-within-tile
  const int halfSel = lane >> 4;        // K-offset selector per WMMA layout
  const int qb      = blockIdx.x * WPB + wave;
  const int ch      = blockIdx.y;
  const int mStart  = ch * CHUNK;

  // --- cache A operand for all 12 K-chunks (96 VGPRs) ---
  const _Float16* aBase = qn + (size_t)(qb * 16 + col) * D_N + halfSel * 8;
  v16h aReg[KT];
#pragma unroll
  for (int kk = 0; kk < KT; ++kk) {
    v8h lo = *(const v8h*)(aBase + kk * 32);
    v8h hi = *(const v8h*)(aBase + kk * 32 + 16);
    aReg[kk] = __builtin_shufflevector(lo, hi, 0, 1, 2, 3, 4, 5, 6, 7,
                                               8, 9, 10, 11, 12, 13, 14, 15);
  }

  // --- cooperative B-tile fill mapping: 256 threads cover 16 rows x 768 B ---
  const int fr = tid >> 4;   // row 0..15
  const int fc = tid & 15;   // 16B chunk 0..15 (x3 strided -> 48B per thread)

  float best[8];
  int   bidx[8];
#pragma unroll
  for (int j = 0; j < 8; ++j) { best[j] = -3.402823466e+38f; bidx[j] = 0; }

  // Prologue: load tile 0 into registers.
  v8h pre[3];
  {
    const _Float16* src = mn + (size_t)(mStart + fr) * D_N + fc * 8;
#pragma unroll
    for (int i = 0; i < 3; ++i) pre[i] = *(const v8h*)(src + i * 128);
  }

  int cur = 0;
  for (int t = 0; t < TILES; ++t) {
    // Commit tile t (in registers) to LDS buffer `cur`.
    {
      _Float16* dst = &Bsh[cur][fr * PITCH + fc * 8];
#pragma unroll
      for (int i = 0; i < 3; ++i) *(v8h*)(dst + i * 128) = pre[i];
    }
    // Issue global loads for tile t+1 (latency hidden under compute of t).
    if (t + 1 < TILES) {
      const _Float16* src =
          mn + (size_t)(mStart + (t + 1) * 16 + fr) * D_N + fc * 8;
#pragma unroll
      for (int i = 0; i < 3; ++i) pre[i] = *(const v8h*)(src + i * 128);
    }
    __syncthreads();   // tile t visible to all waves; prior readers done

    // Compute: 12 WMMAs against LDS-resident B tile.
    const _Float16* bRow = &Bsh[cur][col * PITCH + halfSel * 8];
    v8f acc = {};
#pragma unroll
    for (int kk = 0; kk < KT; ++kk) {
      v8h lo = *(const v8h*)(bRow + kk * 32);
      v8h hi = *(const v8h*)(bRow + kk * 32 + 16);
      v16h b = __builtin_shufflevector(lo, hi, 0, 1, 2, 3, 4, 5, 6, 7,
                                               8, 9, 10, 11, 12, 13, 14, 15);
      acc = __builtin_amdgcn_wmma_f32_16x16x32_f16(
          false, aReg[kk], false, b, (short)0, acc, false, false);
    }

    // acc[j]: sims[qb*16 + j + 8*halfSel][mStart + t*16 + col]
    const int myM = mStart + t * 16 + col;
#pragma unroll
    for (int j = 0; j < 8; ++j) {
      float v = acc[j];
      if (v > best[j]) { best[j] = v; bidx[j] = myM; }
    }

    cur ^= 1;
  }

  // Reduce (max, argmax) across the 16 N-lanes of each half-group
  // (xor masks 8..1 stay within lanes sharing lane>>4).
#pragma unroll
  for (int j = 0; j < 8; ++j) {
    float v  = best[j];
    int   id = bidx[j];
#pragma unroll
    for (int m = 8; m >= 1; m >>= 1) {
      float ov  = __shfl_xor(v, m, 32);
      int   oid = __shfl_xor(id, m, 32);
      if (ov > v || (ov == v && oid < id)) { v = ov; id = oid; }
    }
    if (col == 0) {
      const int q = qb * 16 + j + 8 * halfSel;
      pMax[(size_t)q * SPLIT + ch] = v;
      pIdx[(size_t)q * SPLIT + ch] = id;
    }
  }
}

// ---------------------------------------------------------------------------
// Kernel 3: per-query reduction over SPLIT partials + label gather / fallback.
// Grid: Q blocks of 128 threads (= C).
// ---------------------------------------------------------------------------
__global__ __launch_bounds__(128) void finalize_predictions(
    const float* __restrict__ pMax, const int* __restrict__ pIdx,
    const float* __restrict__ labels, float* __restrict__ out) {
  __shared__ float sv[SPLIT];
  __shared__ int   si[SPLIT];
  const int q = blockIdx.x;
  const int t = threadIdx.x;

  if (t < SPLIT) {
    sv[t] = pMax[(size_t)q * SPLIT + t];
    si[t] = pIdx[(size_t)q * SPLIT + t];
  }
  __syncthreads();

  for (int s = SPLIT / 2; s >= 1; s >>= 1) {
    if (t < s) {
      float ov = sv[t + s];
      int   oi = si[t + s];
      if (ov > sv[t] || (ov == sv[t] && oi < si[t])) { sv[t] = ov; si[t] = oi; }
    }
    __syncthreads();
  }

  const float mx = sv[0];
  const int   mi = si[0];
  out[(size_t)q * C_N + t] =
      (mx > THRESH) ? labels[(size_t)mi * C_N + t] : (1.0f / C_N);
}

// ---------------------------------------------------------------------------
extern "C" void kernel_launch(void* const* d_in, const int* in_sizes, int n_in,
                              void* d_out, int out_size, void* d_ws, size_t ws_size,
                              hipStream_t stream) {
  const float* q_emb   = (const float*)d_in[0];   // [4096, 384]
  const float* m_emb   = (const float*)d_in[1];   // [65536, 384]
  const float* labels  = (const float*)d_in[2];   // [65536, 128]
  float*       out     = (float*)d_out;           // [4096, 128]

  // Workspace layout (all region sizes are 256B multiples):
  char* p = (char*)d_ws;
  _Float16* qn   = (_Float16*)p;  p += (size_t)Q_N * D_N * sizeof(_Float16);  // 3 MB
  _Float16* mn   = (_Float16*)p;  p += (size_t)M_N * D_N * sizeof(_Float16);  // 48 MB
  float*    pMax = (float*)p;     p += (size_t)Q_N * SPLIT * sizeof(float);   // 1 MB
  int*      pIdx = (int*)p;                                                   // 1 MB

  // 1) normalize to f16
  normalize_rows_f16<<<Q_N / 4, 128, 0, stream>>>(q_emb, qn, Q_N);
  normalize_rows_f16<<<M_N / 4, 128, 0, stream>>>(m_emb, mn, M_N);

  // 2) fused WMMA GEMM + max/argmax partials (LDS-shared B tiles)
  dim3 grid(Q_N / QPB, SPLIT);
  sims_argmax_wmma<<<grid, 32 * WPB, 0, stream>>>(qn, mn, pMax, pIdx);

  // 3) final reduce + gather
  finalize_predictions<<<Q_N, 128, 0, stream>>>(pMax, pIdx, labels, out);
}
